// MutiHeadAttention_80161269612931
// MI455X (gfx1250) — matrix operand order repaired and checked by reading, not compiled, not run on hardware
//
#include <hip/hip_runtime.h>
#include <hip/hip_bf16.h>
#include <cstddef>

// ---------------------------------------------------------------------------
// Performer (FAVOR+) multi-head attention forward for MI455X (gfx1250).
// - All matmuls: v_wmma_f32_16x16x32_f16, wave32, register-blocked.
// - Projection GEMM: 64x128 block tile, 2x2 WMMA tiles per wave, LDS
//   double-buffered; A tiles via Tensor Data Mover (tensor_load_to_lds +
//   s_wait_tensorcnt), B tiles cooperatively transposed to K-innermost.
// - All WMMA fragments are 2x b128 vector loads (K-innermost layouts).
// - ctx GEMM: LDS-staged transposed kp/vh tiles, f32 global atomic accumulate.
// Workspace requirement: ~1.1 GB.
// ---------------------------------------------------------------------------

typedef _Float16 half_t;
typedef __attribute__((ext_vector_type(16))) _Float16 v16h;
typedef __attribute__((ext_vector_type(8)))  _Float16 v8h;
typedef __attribute__((ext_vector_type(8)))  float    v8f;
typedef __attribute__((ext_vector_type(4)))  unsigned v4u;
typedef __attribute__((ext_vector_type(8)))  int      v8i;
typedef __attribute__((ext_vector_type(4)))  int      v4i;

#define B_SZ   8
#define SEQ    4096
#define DMODEL 1024
#define NHEAD  16
#define DHEAD  64
#define BH     (B_SZ * NHEAD)          // 128
#define ROWS   (B_SZ * SEQ)            // 32768
#define MFEAT  266                     // performer nb_features
#define MPAD   288                     // padded to 18 tiles of 16
#define NORMC  0.35355339059327373f    // 64^-0.25
#define DIAGC  0.0625f                 // 0.5 * NORMC^2
#define RATIO  0.061313671f            // 266^-0.5
#define EPSK   1e-4f

#if __has_builtin(__builtin_amdgcn_tensor_load_to_lds)
#define HAVE_TDM 1
#else
#define HAVE_TDM 0
#endif

// ---------------------------------------------------------------------------
// WMMA helpers
// ---------------------------------------------------------------------------
__device__ inline v8f wmma_f16(v16h a, v16h b, v8f c) {
    return __builtin_amdgcn_wmma_f32_16x16x32_f16(false, a, false, b,
                                                  (short)0, c, false, false);
}

// Build a 16-half fragment from two aligned 8-half (b128) vector loads.
__device__ inline v16h frag2(const half_t* p0, const half_t* p1) {
    v8h lo = *(const v8h*)p0;
    v8h hi = *(const v8h*)p1;
    return __builtin_shufflevector(lo, hi, 0, 1, 2, 3, 4, 5, 6, 7,
                                   8, 9, 10, 11, 12, 13, 14, 15);
}

__device__ inline void atomicMaxF(float* addr, float val) {
    unsigned int* ua = (unsigned int*)addr;
    unsigned int old = *ua;
    while (__uint_as_float(old) < val) {
        unsigned int assumed = old;
        old = atomicCAS(ua, assumed, __float_as_uint(val));
        if (old == assumed) break;
    }
}

// ---------------------------------------------------------------------------
// Tensor Data Mover: 2-D tile load (2-byte elements) into LDS.
// D# packing per CDNA5 ISA.
// ---------------------------------------------------------------------------
#if HAVE_TDM
__device__ inline void tdm_load_2d(unsigned ldsOff, const void* gptr,
                                   unsigned tileW, unsigned tileH,
                                   unsigned long long tensorW,
                                   unsigned long long tensorH,
                                   unsigned long long strideElems) {
    unsigned long long ga = (unsigned long long)gptr;
    union { unsigned u[4]; v4u v; } g0;
    g0.u[0] = 1u;                                          // count=1
    g0.u[1] = ldsOff;                                      // lds_addr (bytes)
    g0.u[2] = (unsigned)ga;                                // global_addr[31:0]
    g0.u[3] = (unsigned)((ga >> 32) & 0x1FFFFFFu) | (2u << 30); // hi | type=2

    union { unsigned long long q[4]; v8i v; } g1;
    const unsigned long long dataSize = 1ull;              // 2-byte elements
    g1.q[0] = (dataSize << 16)
            | ((tensorW & 0xFFFFull) << 48);
    g1.q[1] = ((tensorW >> 16) & 0xFFFFull)
            | ((tensorH & 0xFFFFFFFFull) << 16)
            | ((unsigned long long)(tileW & 0xFFFFu) << 48);
    g1.q[2] = (unsigned long long)(tileH & 0xFFFFu)
            | ((strideElems & 0xFFFFFFFFull) << 32);
    g1.q[3] = (strideElems >> 32) & 0xFFFFull;

    v4i z4 = {};
#if defined(__clang_major__) && __clang_major__ >= 23
    v8i z8 = {};
    __builtin_amdgcn_tensor_load_to_lds(g0.v, g1.v, z4, z4, z8, 0);
#else
    __builtin_amdgcn_tensor_load_to_lds(g0.v, g1.v, z4, z4, 0);
#endif
}
#endif

// ---------------------------------------------------------------------------
// Conversion / packing / init helpers
// ---------------------------------------------------------------------------
__global__ void cvt_f16_kernel(const float* __restrict__ src,
                               half_t* __restrict__ dst, size_t n) {
    size_t i = (size_t)blockIdx.x * blockDim.x + threadIdx.x;
    if (i < n) dst[i] = (half_t)src[i];
}

__global__ void zero_f32_kernel(float* __restrict__ p, size_t n) {
    size_t i = (size_t)blockIdx.x * blockDim.x + threadIdx.x;
    if (i < n) p[i] = 0.0f;
}

// projC[m][d] = NORMC * proj[m][d], rows m >= MFEAT zero  (MPAD x DHEAD)
__global__ void projpack_kernel(const float* __restrict__ proj,
                                half_t* __restrict__ projC) {
    int i = blockIdx.x * blockDim.x + threadIdx.x;
    if (i >= MPAD * DHEAD) return;
    int m = i / DHEAD;
    float v = (m < MFEAT) ? NORMC * proj[i] : 0.0f;
    projC[i] = (half_t)v;
}

// ctxC[bh][d][m] = (f16) ctxf[bh][m][d]   (d-major -> K-contiguous B operand)
__global__ void ctx_cvt_kernel(const float* __restrict__ ctxf,
                               half_t* __restrict__ ctxC) {
    int i = blockIdx.x * blockDim.x + threadIdx.x;
    if (i >= BH * MPAD * DHEAD) return;
    int bh = i / (MPAD * DHEAD);
    int r  = i % (MPAD * DHEAD);
    int m  = r / DHEAD, d = r % DHEAD;
    ctxC[((size_t)bh * DHEAD + d) * MPAD + m] = (half_t)ctxf[i];
}

__global__ void init_stab_kernel(float* stab) { *stab = -3.0e38f; }

// ---------------------------------------------------------------------------
// Dense GEMM + bias: out[M,N] = A[M,K](f16) * W[K,N](f16) + bias
// Block: 256 threads = 8 waves; block tile 64(M) x 128(N); each wave owns a
// 32x32 output (2x2 WMMA tiles) -> 4 fragment loads feed 4 WMMAs per k-step.
// LDS (24 KB, double buffered):
//   [0,8K)    A buffers, row-major [64][32]        (4 KB each)
//   [8K,24K)  B buffers, TRANSPOSED [128][32]      (8 KB each, K-contiguous)
// A tiles staged by TDM (fallback: cooperative); B tiles coop-transposed.
// ---------------------------------------------------------------------------
__device__ inline void stage_a_coop(half_t* __restrict__ dA,
                                    const half_t* __restrict__ gA, int lda,
                                    int tid) {
    int r = tid >> 2, c = (tid & 3) * 8;       // 64 rows x 32 cols
    const half_t* s = gA + (size_t)r * lda + c;
    half_t* d = dA + r * 32 + c;
#pragma unroll
    for (int e = 0; e < 8; ++e) d[e] = s[e];
}

__device__ inline void stage_b_transposed(half_t* __restrict__ dBt,
                                          const half_t* __restrict__ gB, int ldb,
                                          int tid) {
    int kr = tid >> 3, c = (tid & 7) * 16;     // read row kr, 16 cols of 128
    const half_t* s = gB + (size_t)kr * ldb + c;
#pragma unroll
    for (int e = 0; e < 16; ++e) dBt[(c + e) * 32 + kr] = s[e];
}

template <bool OUT_F32>
__global__ void gemm_bias_kernel(const half_t* __restrict__ A,
                                 const half_t* __restrict__ W,
                                 const float* __restrict__ bias,
                                 void* __restrict__ out,
                                 int M, int K, int N) {
    extern __shared__ __align__(16) char smem[];

    int tid  = threadIdx.x;
    int lane = tid & 31;
    int wave = tid >> 5;
    int blockRow = blockIdx.y * 64;
    int blockCol = blockIdx.x * 128;
    int rowInBlk = (wave >> 2) * 32;           // 0 / 32
    int colInBlk = (wave & 3) * 32;            // 0..96

    // prologue: stage k-step 0 into buffer 0
#if HAVE_TDM
    if (wave == 0) {
        tdm_load_2d(0u, A + (size_t)blockRow * K, 32, 64, K, M, K);
        __builtin_amdgcn_s_wait_tensorcnt(0);
    }
#else
    stage_a_coop((half_t*)smem, A + (size_t)blockRow * K, K, tid);
#endif
    stage_b_transposed((half_t*)(smem + 8192), W + blockCol, N, tid);
    __syncthreads();

    v8f acc00 = {}, acc01 = {}, acc10 = {}, acc11 = {};
    int ar = lane & 15;
    int ak = (lane >> 4) << 3;
    int bc = lane & 15;
    int bk = (lane >> 4) << 3;

    int cur = 0;
    for (int k0 = 0; k0 < K; k0 += 32) {
        int nxt = cur ^ 1;
        bool hasNext = (k0 + 32) < K;

        if (hasNext) {
#if HAVE_TDM
            if (wave == 0)
                tdm_load_2d((unsigned)(nxt * 4096),
                            A + (size_t)blockRow * K + (k0 + 32), 32, 64, K, M, K);
#else
            stage_a_coop((half_t*)(smem + nxt * 4096),
                         A + (size_t)blockRow * K + (k0 + 32), K, tid);
#endif
            stage_b_transposed((half_t*)(smem + 8192 + nxt * 8192),
                               W + (size_t)(k0 + 32) * N + blockCol, N, tid);
        }
        // compute 2x2 tiles from LDS (all-vector b128 fragment loads)
        {
            const half_t* Ab = (const half_t*)(smem + cur * 4096);
            const half_t* Bt = (const half_t*)(smem + 8192 + cur * 8192);
            const half_t* pa0 = Ab + (rowInBlk + ar) * 32 + ak;
            const half_t* pa1 = Ab + (rowInBlk + 16 + ar) * 32 + ak;
            const half_t* pb0 = Bt + (colInBlk + bc) * 32 + bk;
            const half_t* pb1 = Bt + (colInBlk + 16 + bc) * 32 + bk;
            v16h a0 = frag2(pa0, pa0 + 16);
            v16h a1 = frag2(pa1, pa1 + 16);
            v16h b0 = frag2(pb0, pb0 + 16);
            v16h b1 = frag2(pb1, pb1 + 16);
            acc00 = wmma_f16(a0, b0, acc00);
            acc01 = wmma_f16(a0, b1, acc01);
            acc10 = wmma_f16(a1, b0, acc10);
            acc11 = wmma_f16(a1, b1, acc11);
        }
        if (hasNext) {
#if HAVE_TDM
            if (wave == 0) __builtin_amdgcn_s_wait_tensorcnt(0);
#endif
            __syncthreads();
        }
        cur = nxt;
    }

    // write the 2x2 tiles
#pragma unroll
    for (int mi = 0; mi < 2; ++mi) {
#pragma unroll
        for (int ni = 0; ni < 2; ++ni) {
            const v8f* accp = (mi == 0) ? (ni == 0 ? &acc00 : &acc01)
                                        : (ni == 0 ? &acc10 : &acc11);
            int col   = blockCol + colInBlk + ni * 16 + (lane & 15);
            int rbase = blockRow + rowInBlk + mi * 16 + ((lane >> 4) << 3);
            float bv  = bias[col];
            if (OUT_F32) {
                float* o = (float*)out;
#pragma unroll
                for (int v = 0; v < 8; ++v)
                    o[(size_t)(rbase + v) * N + col] = (*accp)[v] + bv;
            } else {
                half_t* o = (half_t*)out;
#pragma unroll
                for (int v = 0; v < 8; ++v)
                    o[(size_t)(rbase + v) * N + col] = (half_t)((*accp)[v] + bv);
            }
        }
    }
}

// ---------------------------------------------------------------------------
// diag[bh*SEQ + n] = 0.5 * NORMC^2 * sum_d Xh[b*SEQ+n, h*64+d]^2
// ---------------------------------------------------------------------------
__global__ void diag_kernel(const half_t* __restrict__ Xh, float* __restrict__ diag) {
    int idx = blockIdx.x * blockDim.x + threadIdx.x;
    if (idx >= BH * SEQ) return;
    int bh = idx / SEQ, n = idx % SEQ;
    int b = bh / NHEAD, h = bh % NHEAD;
    const half_t* p = Xh + ((size_t)(b * SEQ + n)) * DMODEL + h * DHEAD;
    float s = 0.0f;
#pragma unroll 8
    for (int d = 0; d < DHEAD; ++d) { float x = (float)p[d]; s += x * x; }
    diag[idx] = DIAGC * s;
}

// ---------------------------------------------------------------------------
// phi kernel: data_dash = Xh_head @ projC^T (WMMA), then random feature map.
// A fragments (K=64 -> 2 fragments) hoisted out of the feature-tile loop and
// reused across the wave's 3 tiles. All fragment loads are b128 vectors.
// MODE 0: query (per-row stabilizer) -> out
// MODE 1: K max pass (atomic global max only)
// MODE 2: K phi with global stabilizer -> out
// ---------------------------------------------------------------------------
template <int MODE>
__global__ void phi_kernel(const half_t* __restrict__ Xh,
                           const half_t* __restrict__ projC,
                           const float* __restrict__ diag,
                           const float* __restrict__ stabIn,
                           float* __restrict__ stabOut,
                           half_t* __restrict__ out) {
    __shared__ float dd[16][MPAD];
    __shared__ float rmax[16];

    int bh = blockIdx.y;
    int b = bh / NHEAD, h = bh % NHEAD;
    int nBase = blockIdx.x * 16;
    int lane = threadIdx.x & 31, wave = threadIdx.x >> 5;   // 6 waves

    const half_t* Ap = Xh + ((size_t)(b * SEQ + nBase)) * DMODEL + h * DHEAD;
    int arow = lane & 15;
    int ak   = (lane >> 4) << 3;
    int bk   = (lane >> 4) << 3;

    const half_t* pa = Ap + (size_t)arow * DMODEL + ak;
    v16h a0 = frag2(pa, pa + 16);            // K = 0..31
    v16h a1 = frag2(pa + 32, pa + 48);       // K = 32..63

    for (int t = wave; t < MPAD / 16; t += 6) {
        v8f acc = {};
        int col = t * 16 + (lane & 15);
        const half_t* Bp = projC + (size_t)col * DHEAD + bk;
        v16h b0 = frag2(Bp, Bp + 16);
        v16h b1 = frag2(Bp + 32, Bp + 48);
        acc = wmma_f16(a0, b0, acc);
        acc = wmma_f16(a1, b1, acc);
        int r0 = (lane >> 4) << 3;
#pragma unroll
        for (int v = 0; v < 8; ++v) dd[r0 + v][col] = acc[v];
    }
    __syncthreads();

    if (MODE != 2) {
        if (threadIdx.x < 16) {
            float mx = -3.0e38f;
            for (int m = 0; m < MFEAT; ++m) mx = fmaxf(mx, dd[threadIdx.x][m]);
            rmax[threadIdx.x] = mx;
            if (MODE == 1) atomicMaxF(stabOut, mx);
        }
        __syncthreads();
    }
    if (MODE == 1) return;

    float stab = (MODE == 2) ? *stabIn : 0.0f;
    int rowG0 = bh * SEQ + nBase;
    for (int idx = threadIdx.x; idx < 16 * MPAD; idx += blockDim.x) {
        int r = idx / MPAD, m = idx % MPAD;
        float s  = (MODE == 0) ? rmax[r] : stab;
        float dg = diag[rowG0 + r];
        float v  = (m < MFEAT) ? RATIO * (__expf(dd[r][m] - dg - s) + EPSK) : 0.0f;
        out[((size_t)(rowG0 + r)) * MPAD + m] = (half_t)v;
    }
}

// ---------------------------------------------------------------------------
// k_sum[bh][m] = sum_n kp[bh][n][m]
// ---------------------------------------------------------------------------
__global__ void ksum_kernel(const half_t* __restrict__ kp, float* __restrict__ ksum) {
    int bh = blockIdx.x, m = threadIdx.x;
    const half_t* p = kp + (size_t)bh * SEQ * MPAD + m;
    float s = 0.0f;
    for (int n = 0; n < SEQ; ++n) s += (float)p[(size_t)n * MPAD];
    ksum[bh * MPAD + m] = s;
}

// d_inv[bh*SEQ+n] = 1 / sum_m qp[n][m] * ksum[m]   (padded cols are zero)
__global__ void dinv_kernel(const half_t* __restrict__ qp,
                            const float* __restrict__ ksum,
                            float* __restrict__ dinv) {
    int idx = blockIdx.x * blockDim.x + threadIdx.x;
    if (idx >= BH * SEQ) return;
    int bh = idx / SEQ;
    const half_t* p  = qp + (size_t)idx * MPAD;
    const float*  ks = ksum + bh * MPAD;
    float s = 0.0f;
#pragma unroll 8
    for (int m = 0; m < MPAD; ++m) s += (float)p[m] * ks[m];
    dinv[idx] = 1.0f / s;
}

// ---------------------------------------------------------------------------
// ctx[bh][m][d] += sum_{n in chunk} kp[bh][n][m] * vh[b][n][h*64+d]
// Grid (4 n-chunks, BH); block = 8 waves, each wave owns 9 of the 72
// (18 m-tiles x 4 d-tiles) WMMA tiles. K-innermost LDS layouts:
//   kpC [MPAD][32]  (A operand), vhC [DHEAD][32] (B operand).
// Partial sums accumulated with global_atomic_add_f32.
// ---------------------------------------------------------------------------
#define CTX_CHUNKS 4
#define CTX_NSEQ   (SEQ / CTX_CHUNKS)   // 1024

__global__ void ctx_kernel(const half_t* __restrict__ kp,
                           const half_t* __restrict__ Vh,
                           float* __restrict__ ctx) {
    extern __shared__ __align__(16) char smem[];
    half_t* kpC = (half_t*)smem;                    // [MPAD][32]
    half_t* vhC = (half_t*)(smem + MPAD * 32 * 2);  // [DHEAD][32]

    int tid = threadIdx.x, lane = tid & 31, wave = tid >> 5;
    int bh = blockIdx.y;
    int b = bh / NHEAD, h = bh % NHEAD;
    int nBase = blockIdx.x * CTX_NSEQ;

    const half_t* kpB = kp + ((size_t)bh * SEQ + nBase) * MPAD;
    const half_t* vB  = Vh + ((size_t)(b * SEQ + nBase)) * DMODEL + h * DHEAD;

    v8f acc[9];
#pragma unroll
    for (int i = 0; i < 9; ++i) acc[i] = (v8f){};

    for (int k0 = 0; k0 < CTX_NSEQ; k0 += 32) {
        __syncthreads();
        // stage kp tile 32 x 288 transposed -> kpC[m][k]
        for (int i = tid; i < 32 * MPAD / 4; i += 256) {
            int r = i / (MPAD / 4), c = (i % (MPAD / 4)) * 4;
            const half_t* s = kpB + (size_t)(k0 + r) * MPAD + c;
#pragma unroll
            for (int e = 0; e < 4; ++e) kpC[(c + e) * 32 + r] = s[e];
        }
        // stage vh tile 32 x 64 transposed -> vhC[d][k]
        for (int i = tid; i < 32 * 64 / 4; i += 256) {
            int r = i / 16, c = (i % 16) * 4;
            const half_t* s = vB + (size_t)(k0 + r) * DMODEL + c;
#pragma unroll
            for (int e = 0; e < 4; ++e) vhC[(c + e) * 32 + r] = s[e];
        }
        __syncthreads();

#pragma unroll
        for (int i = 0; i < 9; ++i) {
            int t  = wave + 8 * i;          // 0..71
            int mt = t % 18, dt = t / 18;
            int mRow  = mt * 16 + (lane & 15);
            int kaoff = (lane >> 4) << 3;
            int col   = dt * 16 + (lane & 15);
            int kboff = (lane >> 4) << 3;
            v16h a  = frag2(kpC + mRow * 32 + kaoff, kpC + mRow * 32 + kaoff + 16);
            v16h bf = frag2(vhC + col * 32 + kboff,  vhC + col * 32 + kboff + 16);
            acc[i] = wmma_f16(a, bf, acc[i]);
        }
    }

#pragma unroll
    for (int i = 0; i < 9; ++i) {
        int t  = wave + 8 * i;
        int mt = t % 18, dt = t / 18;
        float* o = ctx + (size_t)bh * MPAD * DHEAD
                 + (size_t)(mt * 16 + ((lane >> 4) << 3)) * DHEAD
                 + dt * 16 + (lane & 15);
#pragma unroll
        for (int v = 0; v < 8; ++v) atomicAdd(o + (size_t)v * DHEAD, acc[i][v]);
    }
}

// ---------------------------------------------------------------------------
// attn[b][n][h*64+d] = d_inv[n] * sum_m qp[n][m] * ctxC[d][m]
// One wave per 16-row stripe computes all 4 d-tiles: the A (qp) fragment is
// loaded once per k-step and reused 4x. All fragment loads are b128 vectors.
// ---------------------------------------------------------------------------
__global__ void outattn_kernel(const half_t* __restrict__ qp,
                               const half_t* __restrict__ ctxC,
                               const float* __restrict__ dinv,
                               half_t* __restrict__ attn) {
    int lane = threadIdx.x & 31, wave = threadIdx.x >> 5;
    int bh = blockIdx.y;
    int b = bh / NHEAD, h = bh % NHEAD;
    int mt = blockIdx.x * 8 + wave;            // 0..255 row tiles

    const half_t* Ap = qp + ((size_t)bh * SEQ + mt * 16) * MPAD;
    int arow = lane & 15;
    int ak   = (lane >> 4) << 3;
    int bcol = lane & 15;
    int bk   = (lane >> 4) << 3;
    const half_t* Bbase = ctxC + (size_t)bh * DHEAD * MPAD;

    v8f acc[4];
#pragma unroll
    for (int i = 0; i < 4; ++i) acc[i] = (v8f){};

#pragma unroll
    for (int k0 = 0; k0 < MPAD; k0 += 32) {
        const half_t* pa = Ap + (size_t)arow * MPAD + k0 + ak;
        v16h a = frag2(pa, pa + 16);
#pragma unroll
        for (int dt = 0; dt < 4; ++dt) {
            const half_t* pb = Bbase + (size_t)(dt * 16 + bcol) * MPAD + k0 + bk;
            v16h bf = frag2(pb, pb + 16);
            acc[dt] = wmma_f16(a, bf, acc[dt]);
        }
    }
    int rbase = mt * 16 + ((lane >> 4) << 3);
    half_t* o = attn + ((size_t)b * SEQ) * DMODEL + h * DHEAD;
#pragma unroll
    for (int dt = 0; dt < 4; ++dt) {
        int col = dt * 16 + (lane & 15);
#pragma unroll
        for (int v = 0; v < 8; ++v) {
            int r = rbase + v;
            o[(size_t)r * DMODEL + col] = (half_t)(acc[dt][v] * dinv[bh * SEQ + r]);
        }
    }
}

// ---------------------------------------------------------------------------
// Host-side launch
// ---------------------------------------------------------------------------
static inline size_t alignUp(size_t x) { return (x + 255) & ~(size_t)255; }

extern "C" void kernel_launch(void* const* d_in, const int* in_sizes, int n_in,
                              void* d_out, int out_size, void* d_ws, size_t ws_size,
                              hipStream_t stream) {
    const float* q    = (const float*)d_in[0];
    const float* k    = (const float*)d_in[1];
    const float* v    = (const float*)d_in[2];
    const float* Wq   = (const float*)d_in[3];
    const float* bq   = (const float*)d_in[4];
    const float* Wk   = (const float*)d_in[5];
    const float* bk   = (const float*)d_in[6];
    const float* Wv   = (const float*)d_in[7];
    const float* bv   = (const float*)d_in[8];
    const float* Wo   = (const float*)d_in[9];
    const float* bo   = (const float*)d_in[10];
    const float* proj = (const float*)d_in[11];
    float* out = (float*)d_out;

    char* ws = (char*)d_ws;
    size_t off = 0;
    auto take = [&](size_t bytes) { char* p = ws + off; off += alignUp(bytes); return p; };

    const size_t inElems = (size_t)ROWS * DMODEL;
    half_t* q16   = (half_t*)take(inElems * 2);
    half_t* k16   = (half_t*)take(inElems * 2);
    half_t* v16   = (half_t*)take(inElems * 2);
    half_t* Wq16  = (half_t*)take((size_t)DMODEL * DMODEL * 2);
    half_t* Wk16  = (half_t*)take((size_t)DMODEL * DMODEL * 2);
    half_t* Wv16  = (half_t*)take((size_t)DMODEL * DMODEL * 2);
    half_t* Wo16  = (half_t*)take((size_t)DMODEL * DMODEL * 2);
    half_t* projC = (half_t*)take((size_t)MPAD * DHEAD * 2);
    half_t* Qh    = (half_t*)take(inElems * 2);
    half_t* Kh    = (half_t*)take(inElems * 2);
    half_t* Vh    = (half_t*)take(inElems * 2);
    float*  diagQ = (float*)take((size_t)BH * SEQ * 4);
    float*  diagK = (float*)take((size_t)BH * SEQ * 4);
    float*  stab  = (float*)take(256);
    half_t* qp    = (half_t*)take((size_t)BH * SEQ * MPAD * 2);
    half_t* kp    = (half_t*)take((size_t)BH * SEQ * MPAD * 2);
    float*  ksum  = (float*)take((size_t)BH * MPAD * 4);
    float*  dinv  = (float*)take((size_t)BH * SEQ * 4);
    float*  ctxf  = (float*)take((size_t)BH * MPAD * DHEAD * 4);
    half_t* ctxC  = (half_t*)take((size_t)BH * MPAD * DHEAD * 2);
    half_t* attn  = (half_t*)take(inElems * 2);
    (void)ws_size; (void)n_in; (void)in_sizes; (void)out_size;

    // 1) fp32 -> f16 conversions + init
    {
        int thr = 256;
        unsigned blocks = (unsigned)((inElems + thr - 1) / thr);
        cvt_f16_kernel<<<blocks, thr, 0, stream>>>(q, q16, inElems);
        cvt_f16_kernel<<<blocks, thr, 0, stream>>>(k, k16, inElems);
        cvt_f16_kernel<<<blocks, thr, 0, stream>>>(v, v16, inElems);
        size_t wN = (size_t)DMODEL * DMODEL;
        unsigned wb = (unsigned)((wN + thr - 1) / thr);
        cvt_f16_kernel<<<wb, thr, 0, stream>>>(Wq, Wq16, wN);
        cvt_f16_kernel<<<wb, thr, 0, stream>>>(Wk, Wk16, wN);
        cvt_f16_kernel<<<wb, thr, 0, stream>>>(Wv, Wv16, wN);
        cvt_f16_kernel<<<wb, thr, 0, stream>>>(Wo, Wo16, wN);
        projpack_kernel<<<(MPAD * DHEAD + 255) / 256, 256, 0, stream>>>(proj, projC);
        init_stab_kernel<<<1, 1, 0, stream>>>(stab);
        size_t cN = (size_t)BH * MPAD * DHEAD;
        zero_f32_kernel<<<(unsigned)((cN + 255) / 256), 256, 0, stream>>>(ctxf, cN);
    }

    // 2) Q/K/V projections (TDM/LDS double-buffered WMMA GEMM, 2x2 blocked)
    {
        dim3 grid(DMODEL / 128, ROWS / 64);
        gemm_bias_kernel<false><<<grid, 256, 24576, stream>>>(q16, Wq16, bq, Qh, ROWS, DMODEL, DMODEL);
        gemm_bias_kernel<false><<<grid, 256, 24576, stream>>>(k16, Wk16, bk, Kh, ROWS, DMODEL, DMODEL);
        gemm_bias_kernel<false><<<grid, 256, 24576, stream>>>(v16, Wv16, bv, Vh, ROWS, DMODEL, DMODEL);
    }

    // 3) diag terms
    {
        int total = BH * SEQ;
        diag_kernel<<<(total + 255) / 256, 256, 0, stream>>>(Qh, diagQ);
        diag_kernel<<<(total + 255) / 256, 256, 0, stream>>>(Kh, diagK);
    }

    // 4) random feature maps (WMMA + LDS stripe + exp)
    {
        dim3 grid(SEQ / 16, BH);
        phi_kernel<1><<<grid, 192, 0, stream>>>(Kh, projC, diagK, nullptr, stab, nullptr);
        phi_kernel<0><<<grid, 192, 0, stream>>>(Qh, projC, diagQ, nullptr, nullptr, qp);
        phi_kernel<2><<<grid, 192, 0, stream>>>(Kh, projC, diagK, stab, nullptr, kp);
    }

    // 5) k_sum and d_inv
    ksum_kernel<<<BH, MPAD, 0, stream>>>(kp, ksum);
    dinv_kernel<<<(BH * SEQ + 255) / 256, 256, 0, stream>>>(qp, ksum, dinv);

    // 6) ctx = kp^T @ vh (LDS-staged WMMA, f32 atomic accumulate), then f16
    {
        dim3 gctx(CTX_CHUNKS, BH);
        ctx_kernel<<<gctx, 256, 22528, stream>>>(kp, Vh, ctxf);
        size_t cN = (size_t)BH * MPAD * DHEAD;
        ctx_cvt_kernel<<<(unsigned)((cN + 255) / 256), 256, 0, stream>>>(ctxf, ctxC);
    }

    // 7) out = qp @ ctx * d_inv (WMMA, 4 d-tiles per wave)
    {
        dim3 gout(SEQ / 16 / 8, BH);
        outattn_kernel<<<gout, 256, 0, stream>>>(qp, ctxC, dinv, attn);
    }

    // 8) final projection -> fp32 d_out
    {
        dim3 grid(DMODEL / 128, ROWS / 64);
        gemm_bias_kernel<true><<<grid, 256, 24576, stream>>>(attn, Wo16, bo, out, ROWS, DMODEL, DMODEL);
    }
}
